// DenoiseGCN_90220083020457
// MI455X (gfx1250) — compile-verified, matrix operand order checked by reading, not run on hardware
//
#include <hip/hip_runtime.h>
#include <math.h>

// ---------------------------------------------------------------------------
// DenoiseGCN fused kernel for MI455X (gfx1250, wave32, WMMA)
//   B=512, NUM_V=1024, COORD_DIM=2, TDIM=128, HID=256, 4 GCN layers + head
// Strategy: one workgroup = 128 vertices of one sample + halo 4/side (136 rows,
// padded to 144 = 9 WMMA row tiles). h kept in LDS fp16; all GEMMs via
// v_wmma_f32_16x16x32_f16 with fp32 accumulation; weights streamed from L2.
// ---------------------------------------------------------------------------

typedef __attribute__((ext_vector_type(16))) _Float16 v16h;
typedef __attribute__((ext_vector_type(8)))  _Float16 v8h;
typedef __attribute__((ext_vector_type(8)))  float    v8f;

#define BATCH    512
#define NUMV     1024
#define HID      256
#define TDIM     128
#define CHUNK    128
#define HALO     4
#define ROWS     (CHUNK + 2*HALO)   // 136
#define MT       9                  // row tiles (144 padded rows)
#define ROWS_PAD (MT*16)            // 144
#define LDW      256                // fp16 column stride of LDS row buffers
#define NWAVES   16
#define NTHREADS 512

__device__ __forceinline__ float silu_f(float x) {
    return x / (1.0f + __expf(-x));
}

// Build B operand (K=32 x N=16 strip of W, fp32 row-major KxN in memory).
// Layout mirrors the ISA 16-bit operand table: lane<16 -> n=lane, K in
// {0..7, 16..23}; lane>=16 -> same n, K in {8..15, 24..31}. Rows k>=kvalid
// are substituted with 0 (handles the 130-row W0/res0_W padding safely).
__device__ __forceinline__ v16h gather_b(const float* __restrict__ W, int kvalid,
                                         int ktile, int ntile, int lane) {
    const int n     = ntile * 16 + (lane & 15);
    const int khalf = (lane >> 4) ? 8 : 0;
    v16h b;
#pragma unroll
    for (int e = 0; e < 16; ++e) {
        int k = ktile * 32 + khalf + (e < 8 ? e : e + 8);
        float w = (k < kvalid) ? W[k * HID + n] : 0.0f;
        b[e] = (_Float16)w;
    }
    return b;
}

// Load A operand (M=16 x K=32 tile of the fp16 LDS row buffer).
// lane<16 -> m=lane, K {0..7,16..23}; lane>=16 -> same m, K {8..15,24..31}.
__device__ __forceinline__ v16h load_a(const _Float16* __restrict__ buf,
                                       int mtile, int ktile, int lane) {
    const int m     = mtile * 16 + (lane & 15);
    const int kbase = ktile * 32 + ((lane >> 4) ? 8 : 0);
    const _Float16* p = buf + m * LDW + kbase;
    v8h lo = *(const v8h*)(p);
    v8h hi = *(const v8h*)(p + 16);
    v16h a;
#pragma unroll
    for (int e = 0; e < 8; ++e) { a[e] = lo[e]; a[e + 8] = hi[e]; }
    return a;
}

// Per-wave GEMM strip: dst[:, ntile*16 : +16] = src(144 x K) @ W(K x 256).
// B tiles gathered once and reused across all 9 row tiles.
template <int KT>
__device__ __forceinline__ void mm_strip(_Float16* __restrict__ dst,
                                         const _Float16* __restrict__ src,
                                         const float* __restrict__ W, int kvalid,
                                         int ntile, int lane) {
    v16h Bst[KT];
#pragma unroll
    for (int kt = 0; kt < KT; ++kt) Bst[kt] = gather_b(W, kvalid, kt, ntile, lane);

    const int colw  = ntile * 16 + (lane & 15);
    const int rhalf = (lane >> 4) ? 8 : 0;
#pragma unroll
    for (int mt = 0; mt < MT; ++mt) {
        v8f acc = {0.f, 0.f, 0.f, 0.f, 0.f, 0.f, 0.f, 0.f};
#pragma unroll
        for (int kt = 0; kt < KT; ++kt) {
            v16h a = load_a(src, mt, kt, lane);
            acc = __builtin_amdgcn_wmma_f32_16x16x32_f16(
                false, a, false, Bst[kt], (short)0, acc, false, false);
        }
        const int rbase = mt * 16 + rhalf;
#pragma unroll
        for (int r = 0; r < 8; ++r)
            dst[(rbase + r) * LDW + colw] = (_Float16)acc[r];
    }
}

// --------------------------- time embedding --------------------------------
// temb = silu(sinusoidal(t) @ time_W + time_b), one block per batch element.
__global__ __launch_bounds__(TDIM) void temb_kernel(
    const int* __restrict__ t, const float* __restrict__ time_W,
    const float* __restrict__ time_b, float* __restrict__ temb) {
    __shared__ float feat[TDIM];
    const int b = blockIdx.x, j = threadIdx.x;
    const float tv = (float)t[b];
    if (j < 64) {
        float freq = __expf(-logf(10000.0f) * (float)j / 63.0f);
        float ang  = tv * freq;
        feat[j]      = sinf(ang);
        feat[j + 64] = cosf(ang);
    }
    __syncthreads();
    float s = time_b[j];
#pragma unroll 4
    for (int k = 0; k < TDIM; ++k) s += feat[k] * time_W[k * TDIM + j];
    temb[b * TDIM + j] = silu_f(s);
}

// ------------------------------ main fused kernel --------------------------
__global__ __launch_bounds__(NTHREADS) void gcn_main(
    const float* __restrict__ x, const float* __restrict__ tembg,
    const float* __restrict__ W0, const float* __restrict__ b0,
    const float* __restrict__ W1, const float* __restrict__ b1,
    const float* __restrict__ W2, const float* __restrict__ b2,
    const float* __restrict__ W3, const float* __restrict__ b3,
    const float* __restrict__ res0W,
    const float* __restrict__ hW1, const float* __restrict__ hb1,
    const float* __restrict__ hW2, const float* __restrict__ hb2,
    float* __restrict__ out) {
    __shared__ __attribute__((aligned(16))) _Float16 bufA[ROWS_PAD * LDW];
    __shared__ __attribute__((aligned(16))) _Float16 bufB[ROWS_PAD * LDW];

    const int tid    = threadIdx.x;
    const int lane   = tid & 31;
    const int wave   = tid >> 5;            // 0..15 = owned N tile
    const int sample = blockIdx.x >> 3;     // 512 samples
    const int chunk  = blockIdx.x & 7;      // 8 chunks of 128 vertices
    const int vbase  = chunk * CHUNK;
    const float inv3 = 1.0f / 3.0f;

    // ---- build input h = [coords | temb | zero-pad] into bufA ----
    {
        v8h z = {};
        for (int i = tid; i < ROWS_PAD * LDW / 8; i += NTHREADS)
            ((v8h*)bufA)[i] = z;
    }
    __syncthreads();
    for (int i = tid; i < ROWS * TDIM; i += NTHREADS) {
        int r = i >> 7, c = i & (TDIM - 1);
        bufA[r * LDW + 2 + c] = (_Float16)tembg[sample * TDIM + c];
    }
    for (int i = tid; i < ROWS * 2; i += NTHREADS) {
        int r = i >> 1, c = i & 1;
        int v = (vbase + r - HALO + NUMV) & (NUMV - 1);
        bufA[r * LDW + c] = (_Float16)x[sample * 2048 + v * 2 + c];
    }
    __syncthreads();

    const int col   = wave * 16 + (lane & 15);
    const int rhalf = (lane >> 4) ? 8 : 0;

    // ---- layer 0: g = h@W0 -> bufB ; R = h@res0W kept in registers ----
    mm_strip<5>(bufB, bufA, W0, 130, wave, lane);

    v8f accR[MT];
    {
        v16h Bst[5];
#pragma unroll
        for (int kt = 0; kt < 5; ++kt) Bst[kt] = gather_b(res0W, 130, kt, wave, lane);
#pragma unroll
        for (int mt = 0; mt < MT; ++mt) {
            v8f acc = {0.f, 0.f, 0.f, 0.f, 0.f, 0.f, 0.f, 0.f};
#pragma unroll
            for (int kt = 0; kt < 5; ++kt) {
                v16h a = load_a(bufA, mt, kt, lane);
                acc = __builtin_amdgcn_wmma_f32_16x16x32_f16(
                    false, a, false, Bst[kt], (short)0, acc, false, false);
            }
            accR[mt] = acc;
        }
    }
    __syncthreads();  // bufB complete; all bufA reads complete

#pragma unroll
    for (int mt = 0; mt < MT; ++mt) {
        const int rbase = mt * 16 + rhalf;
#pragma unroll
        for (int r = 0; r < 8; ++r) {
            int row = rbase + r;
            int rp = (row == 0) ? 0 : row - 1;
            int rn = (row == ROWS_PAD - 1) ? row : row + 1;
            float g  = (float)bufB[row * LDW + col];
            float gm = (float)bufB[rp * LDW + col];
            float gp = (float)bufB[rn * LDW + col];
            float v  = (gm + g + gp) * inv3 + b0[col] + accR[mt][r];
            bufA[row * LDW + col] = (_Float16)silu_f(v);
        }
    }
    __syncthreads();

    // ---- layers 1..3: g = h@W -> bufB ; h = silu(agg(g)+b+h) ----
    const float* Ws[3] = {W1, W2, W3};
    const float* bs[3] = {b1, b2, b3};
    for (int L = 0; L < 3; ++L) {
        mm_strip<8>(bufB, bufA, Ws[L], 256, wave, lane);
        __syncthreads();
#pragma unroll
        for (int mt = 0; mt < MT; ++mt) {
            const int rbase = mt * 16 + rhalf;
#pragma unroll
            for (int r = 0; r < 8; ++r) {
                int row = rbase + r;
                int rp = (row == 0) ? 0 : row - 1;
                int rn = (row == ROWS_PAD - 1) ? row : row + 1;
                float g  = (float)bufB[row * LDW + col];
                float gm = (float)bufB[rp * LDW + col];
                float gp = (float)bufB[rn * LDW + col];
                float res = (float)bufA[row * LDW + col];
                float v = (gm + g + gp) * inv3 + bs[L][col] + res;
                bufA[row * LDW + col] = (_Float16)silu_f(v);
            }
        }
        __syncthreads();
    }

    // ---- head layer 1: h = silu(h @ hW1 + hb1) (no aggregation) ----
    mm_strip<8>(bufB, bufA, hW1, 256, wave, lane);
    __syncthreads();
#pragma unroll
    for (int mt = 0; mt < MT; ++mt) {
        const int rbase = mt * 16 + rhalf;
#pragma unroll
        for (int r = 0; r < 8; ++r) {
            int row = rbase + r;
            float v = (float)bufB[row * LDW + col] + hb1[col];
            bufA[row * LDW + col] = (_Float16)silu_f(v);
        }
    }
    __syncthreads();

    // ---- head layer 2: out = h @ hW2 + hb2 (256 -> 2), valid rows only ----
    for (int i = tid; i < CHUNK * 2; i += NTHREADS) {
        int rv = i >> 1, c = i & 1;
        int row = rv + HALO;
        float s = hb2[c];
#pragma unroll 8
        for (int k = 0; k < HID; ++k)
            s += (float)bufA[row * LDW + k] * hW2[k * 2 + c];
        out[sample * 2048 + (vbase + rv) * 2 + c] = s;
    }
}

// ---------------------------------------------------------------------------
extern "C" void kernel_launch(void* const* d_in, const int* in_sizes, int n_in,
                              void* d_out, int out_size, void* d_ws, size_t ws_size,
                              hipStream_t stream) {
    const float* x      = (const float*)d_in[0];
    const int*   t      = (const int*)d_in[1];
    const float* time_W = (const float*)d_in[2];
    const float* time_b = (const float*)d_in[3];
    const float* W0     = (const float*)d_in[4];
    const float* b0     = (const float*)d_in[5];
    const float* W1     = (const float*)d_in[6];
    const float* b1     = (const float*)d_in[7];
    const float* W2     = (const float*)d_in[8];
    const float* b2     = (const float*)d_in[9];
    const float* W3     = (const float*)d_in[10];
    const float* b3     = (const float*)d_in[11];
    const float* res0W  = (const float*)d_in[12];
    const float* hW1    = (const float*)d_in[13];
    const float* hb1    = (const float*)d_in[14];
    const float* hW2    = (const float*)d_in[15];
    const float* hb2    = (const float*)d_in[16];
    float* out = (float*)d_out;

    float* temb = (float*)d_ws;  // 512*128 floats = 256 KB

    temb_kernel<<<BATCH, TDIM, 0, stream>>>(t, time_W, time_b, temb);

    const int nblocks = BATCH * (NUMV / CHUNK);  // 4096
    gcn_main<<<nblocks, NTHREADS, 0, stream>>>(
        x, temb, W0, b0, W1, b1, W2, b2, W3, b3, res0W, hW1, hb1, hW2, hb2, out);
}
